// EGATConv_20641612825179
// MI455X (gfx1250) — matrix operand order repaired and checked by reading, not compiled, not run on hardware
//
#include <hip/hip_runtime.h>
#include <hip/hip_bf16.h>

typedef __attribute__((ext_vector_type(16))) _Float16 v16h;
typedef __attribute__((ext_vector_type(8)))  _Float16 v8h;
typedef __attribute__((ext_vector_type(8)))  float    v8f;

// ---- monotone float<->uint encoding for atomicMax-based segment max ----
__device__ __forceinline__ unsigned encf(float x) {
  unsigned u = __float_as_uint(x);
  return (u & 0x80000000u) ? ~u : (u | 0x80000000u);
}
__device__ __forceinline__ float decf(unsigned u) {
  return __uint_as_float((u & 0x80000000u) ? (u & 0x7fffffffu) : ~u);
}

// A-operand (16x32 f16) for v_wmma_f32_16x16x32_f16, built from a global f32 row.
// Lane (r, hf): elems 0..7 = row[8*hf+i], elems 8..15 = row[16+8*hf+i]  (ISA 7.12.2)
__device__ __forceinline__ v16h load_a_f32(const float* P, int hf) {
  const float4* q0 = (const float4*)(P + 8 * hf);
  const float4* q1 = (const float4*)(P + 16 + 8 * hf);
  float4 a = q0[0], b = q0[1], c = q1[0], d = q1[1];
  v16h A;
  A[0]  = (_Float16)a.x; A[1]  = (_Float16)a.y; A[2]  = (_Float16)a.z; A[3]  = (_Float16)a.w;
  A[4]  = (_Float16)b.x; A[5]  = (_Float16)b.y; A[6]  = (_Float16)b.z; A[7]  = (_Float16)b.w;
  A[8]  = (_Float16)c.x; A[9]  = (_Float16)c.y; A[10] = (_Float16)c.z; A[11] = (_Float16)c.w;
  A[12] = (_Float16)d.x; A[13] = (_Float16)d.y; A[14] = (_Float16)d.z; A[15] = (_Float16)d.w;
  return A;
}

// ---- weight pre-swizzle: B-operand layout (lane = column; K = 8*(lane>>4)+i, +8 for i>=8) ----
__global__ __launch_bounds__(256) void prep_kernel(
    const float* __restrict__ Wn, const float* __restrict__ We,
    const float* __restrict__ Woe, const float* __restrict__ Won,
    const float* __restrict__ Wa,
    _Float16* __restrict__ swWe, _Float16* __restrict__ swWoe,
    _Float16* __restrict__ swWn, _Float16* __restrict__ swWon,
    float* __restrict__ wsum) {
  int t = blockIdx.x * 256 + threadIdx.x;
  if (t < 80 * 32) {
    int tile = t >> 5, lane = t & 31;
    const float* W; int ncols, tloc; _Float16* out;
    if (tile < 48)      { W = We;  ncols = 128; tloc = tile;      out = swWe;  } // 6kt x 8nt
    else if (tile < 56) { W = Woe; ncols = 32;  tloc = tile - 48; out = swWoe; } // 4kt x 2nt
    else if (tile < 72) { W = Wn;  ncols = 128; tloc = tile - 56; out = swWn;  } // 2kt x 8nt
    else                { W = Won; ncols = 32;  tloc = tile - 72; out = swWon; } // 4kt x 2nt
    int ntiles = ncols >> 4;
    int kt = tloc / ntiles, nt = tloc - kt * ntiles;
    int c = lane & 15, hf = lane >> 4;
    _Float16* o = out + (tloc * 32 + lane) * 16;
#pragma unroll
    for (int i = 0; i < 16; ++i)
      o[i] = (_Float16)W[(kt * 32 + (i < 8 ? 8 * hf + i : 16 + 8 * hf + (i - 8))) * ncols + nt * 16 + c];
  } else if (t < 80 * 32 + 32) {
    int fo = t - 80 * 32;
    wsum[fo] = Wa[fo * 4] + Wa[fo * 4 + 1] + Wa[fo * 4 + 2] + Wa[fo * 4 + 3];
  }
}

__global__ __launch_bounds__(256) void init_kernel(float* __restrict__ agg,
                                                   float* __restrict__ den,
                                                   unsigned* __restrict__ menc, int N) {
  int i = blockIdx.x * 256 + threadIdx.x;
  if (i < N * 128) agg[i] = 0.f;
  if (i < N * 4) { den[i] = 0.f; menc[i] = 0x007FFFFFu; /* enc(-inf) */ }
}

// ---- nproj = nfeats @ Wn + bn : one wave per 16-node tile ----
__global__ __launch_bounds__(256) void nodeproj_kernel(
    const float* __restrict__ nfeats, const _Float16* __restrict__ swWn,
    const float* __restrict__ bn, float* __restrict__ nproj, int ntilesN) {
  int wid = threadIdx.x >> 5, lane = threadIdx.x & 31;
  int tile = blockIdx.x * 8 + wid;
  if (tile >= ntilesN) return;
  int r = lane & 15, hf = lane >> 4;
  const float* P = nfeats + (size_t)(tile * 16 + r) * 64;
  v8f acc[8];
#pragma unroll
  for (int nt = 0; nt < 8; ++nt) {
    float b = bn[nt * 16 + r];
#pragma unroll
    for (int g = 0; g < 8; ++g) acc[nt][g] = b;
  }
#pragma unroll
  for (int kc = 0; kc < 2; ++kc) {
    v16h A = load_a_f32(P + kc * 32, hf);
#pragma unroll
    for (int nt = 0; nt < 8; ++nt) {
      v16h B = *(const v16h*)(swWn + ((kc * 8 + nt) * 32 + lane) * 16);
      acc[nt] = __builtin_amdgcn_wmma_f32_16x16x32_f16(false, A, false, B, (short)0, acc[nt], false, false);
    }
  }
#pragma unroll
  for (int nt = 0; nt < 8; ++nt)
#pragma unroll
    for (int g = 0; g < 8; ++g)
      nproj[(size_t)(tile * 16 + g + 8 * hf) * 128 + nt * 16 + r] = acc[nt][g];
}

// ---- big fused edge kernel: f = leaky([h_src|e|h_dst]@We); e_out = f@Woe+boe; logits + seg-max ----
__global__ __launch_bounds__(256) void edge_kernel(
    const float* __restrict__ nfeats, const float* __restrict__ efeats,
    const int* __restrict__ src, const int* __restrict__ dst,
    const _Float16* __restrict__ swWe, const _Float16* __restrict__ swWoe,
    const float* __restrict__ wsum, const float* __restrict__ boe,
    float* __restrict__ eout, float* __restrict__ aex,
    unsigned* __restrict__ menc, int ntilesE) {
  __shared__ __align__(16) _Float16 fst[8][16 * 136];  // padded rows: conflict-free
  int wid = threadIdx.x >> 5, lane = threadIdx.x & 31;
  int tile = blockIdx.x * 8 + wid;
  if (tile >= ntilesE) return;  // wave-uniform
  int r = lane & 15, hf = lane >> 4;
  int e = tile * 16 + r;
  int sn = src[e], dn = dst[e];
  const float* rp0 = nfeats + (size_t)sn * 64;
  const float* rp1 = efeats + (size_t)e * 64;
  const float* rp2 = nfeats + (size_t)dn * 64;

  v8f acc[8] = {};
#pragma unroll
  for (int kc = 0; kc < 6; ++kc) {  // K = 192 = 6 chunks of 32
    const float* P = (kc < 2 ? rp0 : (kc < 4 ? rp1 : rp2)) + (kc & 1) * 32;
    v16h A = load_a_f32(P, hf);
#pragma unroll
    for (int nt = 0; nt < 8; ++nt) {
      v16h B = *(const v16h*)(swWe + ((kc * 8 + nt) * 32 + lane) * 16);
      acc[nt] = __builtin_amdgcn_wmma_f32_16x16x32_f16(false, A, false, B, (short)0, acc[nt], false, false);
    }
  }

  // leaky-relu, stage f (f16) to LDS in row-major [16][136]
  _Float16* my = &fst[wid][0];
#pragma unroll
  for (int nt = 0; nt < 8; ++nt) {
#pragma unroll
    for (int g = 0; g < 8; ++g) {
      float x = acc[nt][g];
      x = x > 0.f ? x : 0.01f * x;
      my[(g + 8 * hf) * 136 + nt * 16 + r] = (_Float16)x;  // D: lane col = r, rows g+8*hf
    }
  }
  // same-wave LDS RAW: DS ops retire in order; just stop compiler reordering
  asm volatile("" ::: "memory");

  // e_out = f @ Woe + boe  (A operand re-read from LDS)
  v8f ea[2];
#pragma unroll
  for (int nt = 0; nt < 2; ++nt) {
    float b = boe[nt * 16 + r];
#pragma unroll
    for (int g = 0; g < 8; ++g) ea[nt][g] = b;
  }
#pragma unroll
  for (int kc = 0; kc < 4; ++kc) {
    v8h lo = *(const v8h*)(my + r * 136 + kc * 32 + 8 * hf);
    v8h hi = *(const v8h*)(my + r * 136 + kc * 32 + 16 + 8 * hf);
    v16h A = __builtin_shufflevector(lo, hi, 0, 1, 2, 3, 4, 5, 6, 7, 8, 9, 10, 11, 12, 13, 14, 15);
#pragma unroll
    for (int nt = 0; nt < 2; ++nt) {
      v16h B = *(const v16h*)(swWoe + ((kc * 2 + nt) * 32 + lane) * 16);
      ea[nt] = __builtin_amdgcn_wmma_f32_16x16x32_f16(false, A, false, B, (short)0, ea[nt], false, false);
    }
  }
#pragma unroll
  for (int nt = 0; nt < 2; ++nt)
#pragma unroll
    for (int g = 0; g < 8; ++g)
      eout[(size_t)(tile * 16 + g + 8 * hf) * 32 + nt * 16 + r] = ea[nt][g];

  // attention logits a[e,h] = sum_fo f[e, h*32+fo]*wsum[fo]; lanes 0-15: heads {0,2}; 16-31: {1,3}
#pragma unroll
  for (int t2 = 0; t2 < 2; ++t2) {
    int h = hf + 2 * t2;
    const v8h* fp = (const v8h*)(my + r * 136 + h * 32);
    float s = 0.f;
#pragma unroll
    for (int jj = 0; jj < 4; ++jj) {
      v8h v = fp[jj];
#pragma unroll
      for (int j = 0; j < 8; ++j) s += (float)v[j] * wsum[jj * 8 + j];
    }
    aex[(size_t)e * 4 + h] = s;
    atomicMax(menc + (size_t)dn * 4 + h, encf(s));
  }
}

__global__ __launch_bounds__(256) void softmax_kernel(
    float* __restrict__ aex, const unsigned* __restrict__ menc,
    float* __restrict__ den, const int* __restrict__ dst, int E) {
  int i = blockIdx.x * 256 + threadIdx.x;
  if (i >= E * 4) return;
  int e = i >> 2, h = i & 3;
  int dn = dst[e];
  float ex = expf(aex[i] - decf(menc[dn * 4 + h]));
  aex[i] = ex;
  atomicAdd(den + dn * 4 + h, ex);
}

__global__ __launch_bounds__(256) void scatter_kernel(
    const float* __restrict__ aex, const float* __restrict__ den,
    const float* __restrict__ nproj, const int* __restrict__ src,
    const int* __restrict__ dst, float* __restrict__ agg, int E) {
  long idx = (long)blockIdx.x * 256 + threadIdx.x;
  if (idx >= (long)E * 128) return;
  int e = (int)(idx >> 7);
  int dloc = (int)(idx & 127);
  int h = dloc >> 5;
  int dn = dst[e], sn = src[e];
  float alpha = aex[e * 4 + h] / den[dn * 4 + h];
  atomicAdd(agg + (size_t)dn * 128 + dloc, alpha * nproj[(size_t)sn * 128 + dloc]);
}

// ---- h_out = (deg>0 ? agg : nproj) @ Won + bon ----
__global__ __launch_bounds__(256) void hout_kernel(
    const float* __restrict__ agg, const float* __restrict__ nproj,
    const float* __restrict__ den, const _Float16* __restrict__ swWon,
    const float* __restrict__ bon, float* __restrict__ hout, int ntilesN) {
  int wid = threadIdx.x >> 5, lane = threadIdx.x & 31;
  int tile = blockIdx.x * 8 + wid;
  if (tile >= ntilesN) return;
  int r = lane & 15, hf = lane >> 4;
  int node = tile * 16 + r;
  const float* P = (den[node * 4] > 0.f ? agg : nproj) + (size_t)node * 128;
  v8f acc[2];
#pragma unroll
  for (int nt = 0; nt < 2; ++nt) {
    float b = bon[nt * 16 + r];
#pragma unroll
    for (int g = 0; g < 8; ++g) acc[nt][g] = b;
  }
#pragma unroll
  for (int kc = 0; kc < 4; ++kc) {
    v16h A = load_a_f32(P + kc * 32, hf);
#pragma unroll
    for (int nt = 0; nt < 2; ++nt) {
      v16h B = *(const v16h*)(swWon + ((kc * 2 + nt) * 32 + lane) * 16);
      acc[nt] = __builtin_amdgcn_wmma_f32_16x16x32_f16(false, A, false, B, (short)0, acc[nt], false, false);
    }
  }
#pragma unroll
  for (int nt = 0; nt < 2; ++nt)
#pragma unroll
    for (int g = 0; g < 8; ++g)
      hout[(size_t)(tile * 16 + g + 8 * hf) * 32 + nt * 16 + r] = acc[nt][g];
}

extern "C" void kernel_launch(void* const* d_in, const int* in_sizes, int n_in,
                              void* d_out, int out_size, void* d_ws, size_t ws_size,
                              hipStream_t stream) {
  const float* nfeats = (const float*)d_in[0];
  const float* efeats = (const float*)d_in[1];
  const int*   src    = (const int*)d_in[2];
  const int*   dst    = (const int*)d_in[3];
  const float* Wn     = (const float*)d_in[4];
  const float* bn     = (const float*)d_in[5];
  const float* We     = (const float*)d_in[6];
  const float* Wa     = (const float*)d_in[7];
  const float* Won    = (const float*)d_in[8];
  const float* bon    = (const float*)d_in[9];
  const float* Woe    = (const float*)d_in[10];
  const float* boe    = (const float*)d_in[11];

  const int N = in_sizes[0] / 64;   // 50000
  const int E = in_sizes[2];        // 800000

  char* ws = (char*)d_ws;
  size_t off = 0;
  auto take = [&](size_t bytes) {
    char* p = ws + off;
    off = (off + bytes + 255) & ~(size_t)255;
    return p;
  };
  float*    wsum  = (float*)take(32 * 4);
  _Float16* swWe  = (_Float16*)take(48 * 512 * 2);
  _Float16* swWoe = (_Float16*)take(8 * 512 * 2);
  _Float16* swWn  = (_Float16*)take(16 * 512 * 2);
  _Float16* swWon = (_Float16*)take(8 * 512 * 2);
  float*    nproj = (float*)take((size_t)N * 128 * 4);
  float*    aex   = (float*)take((size_t)E * 4 * 4);
  unsigned* menc  = (unsigned*)take((size_t)N * 4 * 4);
  float*    den   = (float*)take((size_t)N * 4 * 4);
  float*    agg   = (float*)take((size_t)N * 128 * 4);

  float* hout = (float*)d_out;
  float* eout = (float*)d_out + (size_t)N * 32;

  int ntilesN = (N + 15) / 16, ntilesE = (E + 15) / 16;

  prep_kernel<<<11, 256, 0, stream>>>(Wn, We, Woe, Won, Wa, swWe, swWoe, swWn, swWon, wsum);
  init_kernel<<<(N * 128 + 255) / 256, 256, 0, stream>>>(agg, den, menc, N);
  nodeproj_kernel<<<(ntilesN + 7) / 8, 256, 0, stream>>>(nfeats, swWn, bn, nproj, ntilesN);
  edge_kernel<<<(ntilesE + 7) / 8, 256, 0, stream>>>(nfeats, efeats, src, dst, swWe, swWoe,
                                                     wsum, boe, eout, aex, menc, ntilesE);
  softmax_kernel<<<(E * 4 + 255) / 256, 256, 0, stream>>>(aex, menc, den, dst, E);
  scatter_kernel<<<(int)(((long)E * 128 + 255) / 256), 256, 0, stream>>>(aex, den, nproj, src, dst, agg, E);
  hout_kernel<<<(ntilesN + 7) / 8, 256, 0, stream>>>(agg, nproj, den, swWon, bon, hout, ntilesN);
}
